// HeterogeneousGraphAttentionNetwork_18030272709030
// MI455X (gfx1250) — compile-verified, matrix operand order checked by reading, not run on hardware
//
#include <hip/hip_runtime.h>
#include <math.h>

// ---------------------------------------------------------------------------
// Problem constants (match reference)
// ---------------------------------------------------------------------------
static constexpr int kN   = 50000;   // nodes
static constexpr int kE   = 800000;  // edges
static constexpr int kFN  = 128;     // node feature dim
static constexpr int kFE  = 64;      // edge feature dim
static constexpr int kHID = 256;     // hidden dim = H*C
static constexpr int kH   = 8;       // heads
static constexpr int kG   = 64;      // graphs
static constexpr int kL   = 4;       // layers

typedef __attribute__((ext_vector_type(16))) _Float16 v16h;
typedef __attribute__((ext_vector_type(8)))  _Float16 v8h;
typedef __attribute__((ext_vector_type(8)))  float    v8f;

union V16U { v16h v; v8h h[2]; };

#define DEV __device__ __forceinline__

// Monotonic float <-> uint mapping so float max lowers to native u32 atomics
DEV unsigned encf(float f) {
  int i = __float_as_int(f);
  return (i < 0) ? ~(unsigned)i : ((unsigned)i | 0x80000000u);
}
DEV float decf(unsigned u) {
  int i = (u & 0x80000000u) ? (int)(u & 0x7FFFFFFFu) : (int)~u;
  return __int_as_float(i);
}
DEV float leaky(float v) { return v >= 0.f ? v : 0.2f * v; }

// 16-bit WMMA operand fragment: two contiguous 16B loads per lane
struct Frag { v8h lo, hi; };
DEV Frag load_frag(const _Float16* p) {
  Frag f;
  f.lo = *(const v8h*)p;
  f.hi = *(const v8h*)(p + 16);
  return f;
}
DEV v16h frag_v(Frag f) { V16U u; u.h[0] = f.lo; u.h[1] = f.hi; return u.v; }

#define WMMA_F16(ACC, AV, BV) \
  (ACC) = __builtin_amdgcn_wmma_f32_16x16x32_f16(false, (AV), false, (BV), (short)0, (ACC), false, false)

// ---------------------------------------------------------------------------
// Elementwise helpers
// ---------------------------------------------------------------------------
__global__ void k_f32_to_f16(const float* __restrict__ src, _Float16* __restrict__ dst, int n) {
  int i = blockIdx.x * blockDim.x + threadIdx.x;
  if (i < n) dst[i] = (_Float16)src[i];
}
__global__ void k_fill_f32(float* __restrict__ p, float v, int n) {
  int i = blockIdx.x * blockDim.x + threadIdx.x;
  if (i < n) p[i] = v;
}
__global__ void k_fill_u32(unsigned* __restrict__ p, unsigned v, int n) {
  int i = blockIdx.x * blockDim.x + threadIdx.x;
  if (i < n) p[i] = v;
}

// ---------------------------------------------------------------------------
// Generic WMMA GEMM:  out[M,Nt] = act( A[M,K](f16) * W[Nt,K]^T(f16) + bias )
// One wave per 32x64 tile (two 16-row M-tiles share each B fragment -> 2x
// arithmetic intensity on the streamed operand), double-buffered fragments so
// global_load_b128 of step k+1 overlap the v_wmma of step k, plus
// global_prefetch one K-step ahead.
// Optional fused per-head reduction: outAtt[m,h] = sum_c out[m,h*32+c]*att[h,c]
// through a 32x64 LDS tile (avoids materializing e_proj/xd in HBM).
// ---------------------------------------------------------------------------
__global__ void __launch_bounds__(32)
k_wmma_gemm(const _Float16* __restrict__ A, const _Float16* __restrict__ W,
            const float* __restrict__ bias, float* __restrict__ outF,
            _Float16* __restrict__ outH, const float* __restrict__ att,
            float* __restrict__ outAtt, int M, int K, int Nt, int act)
{
  __shared__ float lds[32 * 64];
  const int lane = threadIdx.x;           // wave32
  const int m0   = blockIdx.x * 32;       // first 16-row tile
  const int m1   = (m0 + 16 + 16 <= M) ? (m0 + 16) : m0;  // clamp tail tile
  const int n0   = blockIdx.y * 64;
  const int lr   = lane & 15;
  const int koff = (lane < 16) ? 0 : 8;   // 16-bit fragment K layout
  const _Float16* Arow0 = A + (size_t)(m0 + lr) * K + koff;
  const _Float16* Arow1 = A + (size_t)(m1 + lr) * K + koff;
  const _Float16* Wc[4];
#pragma unroll
  for (int nb = 0; nb < 4; ++nb)
    Wc[nb] = W + (size_t)(n0 + nb * 16 + lr) * K + koff;

  v8f acc[2][4] = {};

  // ---- prologue: load K-step 0 ----
  Frag a0 = load_frag(Arow0);
  Frag a1 = load_frag(Arow1);
  Frag fb[4];
#pragma unroll
  for (int nb = 0; nb < 4; ++nb) fb[nb] = load_frag(Wc[nb]);

  // ---- pipelined main loop ----
  for (int kk = 32; kk < K; kk += 32) {
    __builtin_prefetch(Arow0 + kk + 32, 0, 3);
    __builtin_prefetch(Arow1 + kk + 32, 0, 3);
#pragma unroll
    for (int nb = 0; nb < 4; ++nb) __builtin_prefetch(Wc[nb] + kk + 32, 0, 3);

    Frag na0 = load_frag(Arow0 + kk);
    Frag na1 = load_frag(Arow1 + kk);
    Frag nfb[4];
#pragma unroll
    for (int nb = 0; nb < 4; ++nb) nfb[nb] = load_frag(Wc[nb] + kk);

    const v16h A0 = frag_v(a0), A1 = frag_v(a1);
#pragma unroll
    for (int nb = 0; nb < 4; ++nb) {
      const v16h B = frag_v(fb[nb]);
      WMMA_F16(acc[0][nb], A0, B);
      WMMA_F16(acc[1][nb], A1, B);
    }
    a0 = na0; a1 = na1;
#pragma unroll
    for (int nb = 0; nb < 4; ++nb) fb[nb] = nfb[nb];
  }
  // ---- epilogue compute on last fragments ----
  {
    const v16h A0 = frag_v(a0), A1 = frag_v(a1);
#pragma unroll
    for (int nb = 0; nb < 4; ++nb) {
      const v16h B = frag_v(fb[nb]);
      WMMA_F16(acc[0][nb], A0, B);
      WMMA_F16(acc[1][nb], A1, B);
    }
  }

  // ---- store / fused epilogue ----
  const int rbase = (lane < 16) ? 0 : 8;  // C/D layout: vgpr r -> M=r or r+8
#pragma unroll
  for (int mt = 0; mt < 2; ++mt) {
    const int mbase = (mt == 0) ? m0 : m1;
#pragma unroll
    for (int nb = 0; nb < 4; ++nb) {
      const int col = n0 + nb * 16 + lr;
      const float bv = bias ? bias[col] : 0.0f;
#pragma unroll
      for (int r = 0; r < 8; ++r) {
        float v = acc[mt][nb][r] + bv;
        if (act == 1) v = v > 0.f ? v : 0.f;         // ReLU
        const int orow = mbase + rbase + r;
        const size_t oidx = (size_t)orow * Nt + col;
        if (outF) outF[oidx] = v;
        if (outH) outH[oidx] = (_Float16)v;
        if (outAtt) lds[(mt * 16 + rbase + r) * 64 + nb * 16 + lr] = v;
      }
    }
  }

  if (outAtt) {
    asm volatile("s_wait_dscnt 0" ::: "memory");
    const int lrow = lane;                          // LDS row 0..31
    const int grow = (lrow < 16) ? (m0 + lrow) : (m1 + lrow - 16);
#pragma unroll
    for (int hb = 0; hb < 2; ++hb) {                // 2 heads per 64-col tile
      const int headG = (n0 >> 5) + hb;
      float s = 0.f;
#pragma unroll
      for (int j = 0; j < 32; ++j)
        s += lds[lrow * 64 + hb * 32 + j] * att[headG * 32 + j];
      outAtt[(size_t)grow * kH + headG] = s;
    }
  }
}

// ---------------------------------------------------------------------------
// LayerNorm + LeakyReLU over 256 channels (one block per node)
// ---------------------------------------------------------------------------
__global__ void __launch_bounds__(256)
k_ln_leaky(const float* __restrict__ hin, const float* __restrict__ gam,
           const float* __restrict__ bet, float* __restrict__ hout,
           _Float16* __restrict__ houtH)
{
  __shared__ float red[256];
  const int n = blockIdx.x, c = threadIdx.x;
  const size_t idx = (size_t)n * kHID + c;
  float v = hin[idx];
  red[c] = v; __syncthreads();
  for (int s = 128; s > 0; s >>= 1) { if (c < s) red[c] += red[c + s]; __syncthreads(); }
  float mu = red[0] * (1.0f / kHID);
  __syncthreads();
  float d = v - mu;
  red[c] = d * d; __syncthreads();
  for (int s = 128; s > 0; s >>= 1) { if (c < s) red[c] += red[c + s]; __syncthreads(); }
  float var = red[0] * (1.0f / kHID);
  float o = d * rsqrtf(var + 1e-5f) * gam[c] + bet[c];
  o = leaky(o);
  hout[idx] = o;
  houtH[idx] = (_Float16)o;
}

// ---------------------------------------------------------------------------
// Attention: alpha = leaky(a_src[src]+a_dst[dst]+ealpha); segment max/softmax
// ---------------------------------------------------------------------------
__global__ void k_alpha_max(const int* __restrict__ ei, const float* __restrict__ a_src,
                            const float* __restrict__ a_dst, const float* __restrict__ ealpha,
                            float* __restrict__ alpha, unsigned* __restrict__ menc)
{
  int idx = blockIdx.x * blockDim.x + threadIdx.x;
  if (idx >= kE * kH) return;
  int e = idx >> 3, hh = idx & 7;
  int s = ei[e], d = ei[kE + e];
  float al = a_src[s * kH + hh] + a_dst[d * kH + hh] + ealpha[idx];
  al = leaky(al);
  alpha[idx] = al;
  atomicMax(&menc[d * kH + hh], encf(al));   // native global_atomic_max_u32
}

__global__ void k_alpha_exp(const int* __restrict__ ei, float* __restrict__ alpha,
                            const unsigned* __restrict__ menc, float* __restrict__ denom)
{
  int idx = blockIdx.x * blockDim.x + threadIdx.x;
  if (idx >= kE * kH) return;
  int e = idx >> 3, hh = idx & 7;
  int d = ei[kE + e];
  float ex = expf(alpha[idx] - decf(menc[d * kH + hh]));
  alpha[idx] = ex;
  atomicAdd(&denom[d * kH + hh], ex);        // native global_atomic_add_f32
}

// msg = xs[src]*alpha/denom, scatter-add into agg[dst]  (one block per edge)
__global__ void __launch_bounds__(256)
k_message(const int* __restrict__ ei, const float* __restrict__ xs,
          const float* __restrict__ alpha, const float* __restrict__ denom,
          float* __restrict__ agg)
{
  const int e = blockIdx.x;
  const int c = threadIdx.x;
  const int hh = c >> 5;
  const int s = ei[e], d = ei[kE + e];
  const float coeff = alpha[e * kH + hh] / (denom[d * kH + hh] + 1e-16f);
  atomicAdd(&agg[(size_t)d * kHID + c], xs[(size_t)s * kHID + c] * coeff);
}

// ---------------------------------------------------------------------------
// BatchNorm (batch statistics) + LeakyReLU + residual
// ---------------------------------------------------------------------------
__global__ void __launch_bounds__(256)
k_bn_reduce(const float* __restrict__ agg, const float* __restrict__ gbias,
            float* __restrict__ s1, float* __restrict__ s2)
{
  const int c = threadIdx.x;
  int r0 = blockIdx.x * 128;
  int r1 = r0 + 128; if (r1 > kN) r1 = kN;
  float a = 0.f, b = 0.f;
  const float bv = gbias[c];
  for (int r = r0; r < r1; ++r) {
    float v = agg[(size_t)r * kHID + c] + bv;
    a += v; b += v * v;
  }
  atomicAdd(&s1[c], a);
  atomicAdd(&s2[c], b);
}

__global__ void k_bn_apply(const float* __restrict__ agg, const float* __restrict__ gbias,
                           const float* __restrict__ s1, const float* __restrict__ s2,
                           const float* __restrict__ g, const float* __restrict__ b,
                           const float* __restrict__ hprev, float* __restrict__ hout,
                           _Float16* __restrict__ houtH, int addRes)
{
  int idx = blockIdx.x * blockDim.x + threadIdx.x;
  if (idx >= kN * kHID) return;
  int c = idx & 255;
  float mu  = s1[c] * (1.0f / kN);
  float var = s2[c] * (1.0f / kN) - mu * mu;
  float v = agg[idx] + gbias[c];
  v = (v - mu) * rsqrtf(var + 1e-5f) * g[c] + b[c];
  v = leaky(v);
  if (addRes) v += hprev[idx];
  hout[idx] = v;
  houtH[idx] = (_Float16)v;
}

// ---------------------------------------------------------------------------
// Graph pooling (mean + max) and readout prep
// ---------------------------------------------------------------------------
__global__ void k_count(const int* __restrict__ batch, float* __restrict__ cnt) {
  int n = blockIdx.x * blockDim.x + threadIdx.x;
  if (n < kN) atomicAdd(&cnt[batch[n]], 1.0f);
}
__global__ void k_pool(const int* __restrict__ batch, const float* __restrict__ h,
                       float* __restrict__ gsum, unsigned* __restrict__ gmax)
{
  int idx = blockIdx.x * blockDim.x + threadIdx.x;
  if (idx >= kN * kHID) return;
  int n = idx >> 8, c = idx & 255;
  int g = batch[n];
  float v = h[idx];
  atomicAdd(&gsum[g * kHID + c], v);
  atomicMax(&gmax[g * kHID + c], encf(v));
}
__global__ void k_gcat(const float* __restrict__ gsum, const unsigned* __restrict__ gmax,
                       const float* __restrict__ cnt, _Float16* __restrict__ gcat)
{
  int idx = blockIdx.x * blockDim.x + threadIdx.x;
  if (idx >= kG * 2 * kHID) return;
  int g = idx / (2 * kHID), c = idx % (2 * kHID);
  float v = (c < kHID) ? gsum[g * kHID + c] / fmaxf(cnt[g], 1.0f)
                       : decf(gmax[g * kHID + (c - kHID)]);
  gcat[idx] = (_Float16)v;
}

// ---------------------------------------------------------------------------
// Host orchestration
// ---------------------------------------------------------------------------
extern "C" void kernel_launch(void* const* d_in, const int* in_sizes, int n_in,
                              void* d_out, int out_size, void* d_ws, size_t ws_size,
                              hipStream_t stream) {
  (void)in_sizes; (void)n_in; (void)out_size; (void)ws_size;
  const float* x          = (const float*)d_in[0];
  const int*   ei         = (const int*)  d_in[1];
  const float* edge_attr  = (const float*)d_in[2];
  const int*   batch      = (const int*)  d_in[3];
  const float* emb_W      = (const float*)d_in[4];
  const float* emb_b      = (const float*)d_in[5];
  const float* ln_g       = (const float*)d_in[6];
  const float* ln_b       = (const float*)d_in[7];
  const float* edge_W1    = (const float*)d_in[8];
  const float* edge_b1    = (const float*)d_in[9];
  const float* edge_W2    = (const float*)d_in[10];
  const float* edge_b2    = (const float*)d_in[11];
  const float* lin_src_W  = (const float*)d_in[12];
  const float* lin_dst_W  = (const float*)d_in[13];
  const float* lin_edge_W = (const float*)d_in[14];
  const float* att_src    = (const float*)d_in[15];
  const float* att_dst    = (const float*)d_in[16];
  const float* att_edge   = (const float*)d_in[17];
  const float* gat_bias   = (const float*)d_in[18];
  const float* bn_g       = (const float*)d_in[19];
  const float* bn_b       = (const float*)d_in[20];
  const float* out_W1     = (const float*)d_in[21];
  const float* out_b1     = (const float*)d_in[22];
  const float* out_W2     = (const float*)d_in[23];
  const float* out_b2     = (const float*)d_in[24];

  // Bump allocator over workspace (256B aligned)
  char* ws = (char*)d_ws; size_t off = 0;
  auto alloc = [&](size_t bytes) -> void* {
    void* p = ws + off;
    off = (off + bytes + 255) & ~(size_t)255;
    return p;
  };

  // fp32 buffers (hA..hB and xs..agg are contiguous pairs; the big fp16 edge
  // scratch aliases them since edge MLP runs before any node-side use)
  float* hA   = (float*)alloc((size_t)kN * kHID * 4);
  float* hB   = (float*)alloc((size_t)kN * kHID * 4);
  float* xs   = (float*)alloc((size_t)kN * kHID * 4);
  float* agg  = (float*)alloc((size_t)kN * kHID * 4);
  _Float16* h_h  = (_Float16*)alloc((size_t)kN * kHID * 2);
  _Float16* x_h  = (_Float16*)alloc((size_t)kN * kFN * 2);
  _Float16* ea_h = (_Float16*)alloc((size_t)kE * 128 * 2);
  float* alpha  = (float*)alloc((size_t)kE * kH * 4);
  float* ealpha = (float*)alloc((size_t)kE * kH * 4);
  float* a_src  = (float*)alloc((size_t)kN * kH * 4);
  float* a_dst  = (float*)alloc((size_t)kN * kH * 4);
  unsigned* menc = (unsigned*)alloc((size_t)kN * kH * 4);
  float* denom  = (float*)alloc((size_t)kN * kH * 4);
  float* s12    = (float*)alloc(512 * 4);
  float* gsum   = (float*)alloc((size_t)kG * kHID * 4);
  unsigned* gmax = (unsigned*)alloc((size_t)kG * kHID * 4);
  float* cnt    = (float*)alloc(kG * 4);
  _Float16* gcat_h = (_Float16*)alloc((size_t)kG * 2 * kHID * 2);
  _Float16* g1_h   = (_Float16*)alloc((size_t)kG * kHID * 2);
  // fp16 weights
  _Float16* embW_h    = (_Float16*)alloc(256 * 128 * 2);
  _Float16* ew1_h     = (_Float16*)alloc(64 * 64 * 2);
  _Float16* ew2_h     = (_Float16*)alloc(128 * 64 * 2);
  _Float16* linsrc_h  = (_Float16*)alloc((size_t)kL * 256 * 256 * 2);
  _Float16* lindst_h  = (_Float16*)alloc((size_t)kL * 256 * 256 * 2);
  _Float16* linedge_h = (_Float16*)alloc((size_t)kL * 256 * 128 * 2);
  _Float16* outW1_h   = (_Float16*)alloc(256 * 512 * 2);
  _Float16* outW2_h   = (_Float16*)alloc(256 * 256 * 2);
  // fp16 edge scratch aliased onto hA+hB / xs+agg (dead at that time)
  _Float16* eattr_h = (_Float16*)hA;   // [E,64]  = 102.4MB over hA+hB
  _Float16* t1_h    = (_Float16*)xs;   // [E,64]  = 102.4MB over xs+agg

  auto cdiv = [](long long a, long long b) { return (unsigned)((a + b - 1) / b); };

  // ---- fp32 -> fp16 conversions ----
  k_f32_to_f16<<<cdiv((long long)kN * kFN, 256), 256, 0, stream>>>(x, x_h, kN * kFN);
  k_f32_to_f16<<<cdiv((long long)kE * kFE, 256), 256, 0, stream>>>(edge_attr, eattr_h, kE * kFE);
  k_f32_to_f16<<<cdiv(256 * 128, 256), 256, 0, stream>>>(emb_W, embW_h, 256 * 128);
  k_f32_to_f16<<<cdiv(64 * 64, 256), 256, 0, stream>>>(edge_W1, ew1_h, 64 * 64);
  k_f32_to_f16<<<cdiv(128 * 64, 256), 256, 0, stream>>>(edge_W2, ew2_h, 128 * 64);
  k_f32_to_f16<<<cdiv(kL * 256 * 256, 256), 256, 0, stream>>>(lin_src_W, linsrc_h, kL * 256 * 256);
  k_f32_to_f16<<<cdiv(kL * 256 * 256, 256), 256, 0, stream>>>(lin_dst_W, lindst_h, kL * 256 * 256);
  k_f32_to_f16<<<cdiv(kL * 256 * 128, 256), 256, 0, stream>>>(lin_edge_W, linedge_h, kL * 256 * 128);
  k_f32_to_f16<<<cdiv(256 * 512, 256), 256, 0, stream>>>(out_W1, outW1_h, 256 * 512);
  k_f32_to_f16<<<cdiv(256 * 256, 256), 256, 0, stream>>>(out_W2, outW2_h, 256 * 256);

  // ---- Edge MLP: relu(ea @ W1^T + b1) @ W2^T + b2 -> ea_h [E,128] (fp16) ----
  k_wmma_gemm<<<dim3(cdiv(kE, 32), 1), 32, 0, stream>>>(
      eattr_h, ew1_h, edge_b1, nullptr, t1_h, nullptr, nullptr, kE, 64, 64, 1);
  k_wmma_gemm<<<dim3(cdiv(kE, 32), 2), 32, 0, stream>>>(
      t1_h, ew2_h, edge_b2, nullptr, ea_h, nullptr, nullptr, kE, 64, 128, 0);

  // ---- Node embedding: GEMM -> LayerNorm -> LeakyReLU ----
  k_wmma_gemm<<<dim3(cdiv(kN, 32), kHID / 64), 32, 0, stream>>>(
      x_h, embW_h, emb_b, hB, nullptr, nullptr, nullptr, kN, kFN, kHID, 0);
  k_ln_leaky<<<kN, 256, 0, stream>>>(hB, ln_g, ln_b, hA, h_h);

  float* hcur = hA;
  float* hoth = hB;

  // ---- GAT layers ----
  for (int i = 0; i < kL; ++i) {
    k_fill_u32<<<cdiv(kN * kH, 256), 256, 0, stream>>>(menc, 0u, kN * kH);
    k_fill_f32<<<cdiv(kN * kH, 256), 256, 0, stream>>>(denom, 0.f, kN * kH);
    k_fill_f32<<<cdiv((long long)kN * kHID, 256), 256, 0, stream>>>(agg, 0.f, kN * kHID);
    k_fill_f32<<<2, 256, 0, stream>>>(s12, 0.f, 512);

    // xs = h @ lin_src^T (store fp32 + fused a_src); a_dst fused only
    k_wmma_gemm<<<dim3(cdiv(kN, 32), kHID / 64), 32, 0, stream>>>(
        h_h, linsrc_h + (size_t)i * 256 * 256, nullptr, xs, nullptr,
        att_src + i * kH * 32, a_src, kN, kHID, kHID, 0);
    k_wmma_gemm<<<dim3(cdiv(kN, 32), kHID / 64), 32, 0, stream>>>(
        h_h, lindst_h + (size_t)i * 256 * 256, nullptr, nullptr, nullptr,
        att_dst + i * kH * 32, a_dst, kN, kHID, kHID, 0);
    // e_proj never materialized: fused head reduction -> ealpha [E,8]
    k_wmma_gemm<<<dim3(cdiv(kE, 32), kHID / 64), 32, 0, stream>>>(
        ea_h, linedge_h + (size_t)i * 256 * 128, nullptr, nullptr, nullptr,
        att_edge + i * kH * 32, ealpha, kE, 128, kHID, 0);

    k_alpha_max<<<cdiv((long long)kE * kH, 256), 256, 0, stream>>>(
        ei, a_src, a_dst, ealpha, alpha, menc);
    k_alpha_exp<<<cdiv((long long)kE * kH, 256), 256, 0, stream>>>(
        ei, alpha, menc, denom);
    k_message<<<kE, 256, 0, stream>>>(ei, xs, alpha, denom, agg);

    k_bn_reduce<<<cdiv(kN, 128), 256, 0, stream>>>(agg, gat_bias + i * kHID, s12, s12 + 256);
    k_bn_apply<<<cdiv((long long)kN * kHID, 256), 256, 0, stream>>>(
        agg, gat_bias + i * kHID, s12, s12 + 256, bn_g + i * kHID, bn_b + i * kHID,
        hcur, hoth, h_h, (i > 0) ? 1 : 0);
    float* t = hcur; hcur = hoth; hoth = t;
  }

  // ---- Pooling + readout MLP ----
  k_fill_f32<<<cdiv(kG * kHID, 256), 256, 0, stream>>>(gsum, 0.f, kG * kHID);
  k_fill_u32<<<cdiv(kG * kHID, 256), 256, 0, stream>>>(gmax, 0u, kG * kHID);
  k_fill_f32<<<1, 256, 0, stream>>>(cnt, 0.f, kG);
  k_count<<<cdiv(kN, 256), 256, 0, stream>>>(batch, cnt);
  k_pool<<<cdiv((long long)kN * kHID, 256), 256, 0, stream>>>(batch, hcur, gsum, gmax);
  k_gcat<<<cdiv(kG * 2 * kHID, 256), 256, 0, stream>>>(gsum, gmax, cnt, gcat_h);

  k_wmma_gemm<<<dim3(cdiv(kG, 32), kHID / 64), 32, 0, stream>>>(
      gcat_h, outW1_h, out_b1, nullptr, g1_h, nullptr, nullptr, kG, 2 * kHID, kHID, 1);
  k_wmma_gemm<<<dim3(cdiv(kG, 32), kHID / 64), 32, 0, stream>>>(
      g1_h, outW2_h, out_b2, (float*)d_out, nullptr, nullptr, nullptr, kG, kHID, kHID, 0);
}
